// FFiNetConcatModel_71030169141775
// MI455X (gfx1250) — matrix operand order repaired and checked by reading, not compiled, not run on hardware
//
#include <hip/hip_runtime.h>
#include <hip/hip_bf16.h>
#include <math.h>

#define HID 128
#define HEADS 8
#define DH 16
#define EDIM 16
#define LAYERS 3
#define KPAD 288      // 257 padded up to a multiple of 32

typedef __attribute__((ext_vector_type(16))) _Float16 v16h;
typedef __attribute__((ext_vector_type(8)))  _Float16 v8h;
typedef __attribute__((ext_vector_type(4)))  _Float16 v4h;
typedef __attribute__((ext_vector_type(8)))  float    v8f;

// ---------- helpers ----------
__device__ __forceinline__ unsigned fenc(float f) {
  unsigned u = __float_as_uint(f);
  return (u & 0x80000000u) ? ~u : (u | 0x80000000u);   // order-preserving
}
__device__ __forceinline__ float fdec(unsigned u) {
  return (u & 0x80000000u) ? __uint_as_float(u & 0x7FFFFFFFu)
                           : __uint_as_float(~u);
}
__device__ __forceinline__ float eluf(float x) {
  return x > 0.f ? x : (__expf(x) - 1.f);
}

// ---------- geometry ----------
__global__ void edge_geom(const float* __restrict__ pos,
                          const int* __restrict__ ei, float* __restrict__ d, int E) {
  int e = blockIdx.x * blockDim.x + threadIdx.x;
  if (e >= E) return;
  int i0 = ei[e], i1 = ei[E + e];
  float dx = pos[3*i0+0] - pos[3*i1+0] + 1e-8f;
  float dy = pos[3*i0+1] - pos[3*i1+1] + 1e-8f;
  float dz = pos[3*i0+2] - pos[3*i1+2] + 1e-8f;
  d[e] = sqrtf(dx*dx + dy*dy + dz*dz);
}

__global__ void triple_geom(const float* __restrict__ pos,
                            const int* __restrict__ ti, float* __restrict__ ca, int T) {
  int t = blockIdx.x * blockDim.x + threadIdx.x;
  if (t >= T) return;
  int i = ti[t], j = ti[T + t], k = ti[2*T + t];
  float ux = pos[3*i+0]-pos[3*j+0], uy = pos[3*i+1]-pos[3*j+1], uz = pos[3*i+2]-pos[3*j+2];
  float vx = pos[3*k+0]-pos[3*j+0], vy = pos[3*k+1]-pos[3*j+1], vz = pos[3*k+2]-pos[3*j+2];
  float dot = ux*vx + uy*vy + uz*vz;
  float nu = sqrtf(ux*ux+uy*uy+uz*uz), nv = sqrtf(vx*vx+vy*vy+vz*vz);
  ca[t] = dot / (nu*nv + 1e-8f);
}

__global__ void quad_geom(const float* __restrict__ pos,
                          const int* __restrict__ qi, float* __restrict__ cd, int Q) {
  int q = blockIdx.x * blockDim.x + threadIdx.x;
  if (q >= Q) return;
  int p0 = qi[q], p1 = qi[Q+q], p2 = qi[2*Q+q], p3 = qi[3*Q+q];
  float b1x = pos[3*p1+0]-pos[3*p0+0], b1y = pos[3*p1+1]-pos[3*p0+1], b1z = pos[3*p1+2]-pos[3*p0+2];
  float b2x = pos[3*p2+0]-pos[3*p1+0], b2y = pos[3*p2+1]-pos[3*p1+1], b2z = pos[3*p2+2]-pos[3*p1+2];
  float b3x = pos[3*p3+0]-pos[3*p2+0], b3y = pos[3*p3+1]-pos[3*p2+1], b3z = pos[3*p3+2]-pos[3*p2+2];
  float n1x = b1y*b2z - b1z*b2y, n1y = b1z*b2x - b1x*b2z, n1z = b1x*b2y - b1y*b2x;
  float n2x = b2y*b3z - b2z*b3y, n2y = b2z*b3x - b2x*b3z, n2z = b2x*b3y - b2y*b3x;
  float dot = n1x*n2x + n1y*n2y + n1z*n2z;
  float nn1 = sqrtf(n1x*n1x+n1y*n1y+n1z*n1z), nn2 = sqrtf(n2x*n2x+n2y*n2y+n2z*n2z);
  cd[q] = dot / (nn1*nn2 + 1e-8f);
}

// ---------- WMMA GEMM: C[MxN] = act(A[MxK] * B[KxN] + bias) ----------
// Block = 128 threads (4 waves); tile = 64 rows x 16 cols; K multiple of 32;
// N multiple of 16. LDS-staged, f16 in / f32 acc, register double-buffered,
// branch-free staging (clamped addresses + value selects).
#define TM 64
#define ASTR 48   // padded row stride (halfs) to spread LDS banks

struct GTile {
  float4 a[4];
  float4 b;
};

__global__ __launch_bounds__(128) void wmma_gemm(
    const float* __restrict__ A, int lda,
    const float* __restrict__ B, int ldb,
    float* __restrict__ C, int ldc,
    int M, int K,
    const float* __restrict__ bias, int act) {
  __shared__ __align__(16) _Float16 Alds[TM][ASTR];
  __shared__ __align__(16) _Float16 Blds[16][ASTR];
  const int tid = threadIdx.x;
  const int wave = tid >> 5;
  const int lane = tid & 31;
  const int hi = lane >> 4;        // 0: lanes 0-15, 1: lanes 16-31
  const int lane16 = lane & 15;
  const int gm = blockIdx.y * TM;
  const int gn = blockIdx.x * 16;
  const int ml = wave * 16 + lane16;

  // per-thread staging coordinates (fixed across k-chunks)
  const int arow = tid >> 3;              // 0..15 (x4 reps -> 0..63)
  const int acg  = (tid & 7) * 4;         // 0,4,...,28
  const int bkr  = tid >> 2;              // 0..31
  const int bcg  = (tid & 3) * 4;         // 0,4,8,12
  bool avalid[4];
#pragma unroll
  for (int r = 0; r < 4; ++r) avalid[r] = (gm + r * 16 + arow) < M;

  // branch-free tile fetch: clamped row so loads are always legal,
  // zeros selected in at conversion time (keeps loads unconditional).
  auto fetch = [&](int kc, GTile& t) {
#pragma unroll
    for (int r = 0; r < 4; ++r) {
      int rg = gm + r * 16 + arow;
      int rc = rg < M ? rg : M - 1;
      t.a[r] = *(const float4*)(A + (long long)rc * lda + kc + acg);
    }
    t.b = *(const float4*)(B + (long long)(kc + bkr) * ldb + gn + bcg);
  };

  auto stage = [&](const GTile& t) {
#pragma unroll
    for (int r = 0; r < 4; ++r) {
      float4 a4 = t.a[r];
      if (!avalid[r]) a4 = make_float4(0.f, 0.f, 0.f, 0.f);   // v_cndmask
      v4h hv = { (_Float16)a4.x, (_Float16)a4.y, (_Float16)a4.z, (_Float16)a4.w };
      *(v4h*)&Alds[r * 16 + arow][acg] = hv;
    }
    Blds[bcg + 0][bkr] = (_Float16)t.b.x;
    Blds[bcg + 1][bkr] = (_Float16)t.b.y;
    Blds[bcg + 2][bkr] = (_Float16)t.b.z;
    Blds[bcg + 3][bkr] = (_Float16)t.b.w;
  };

  v8f acc = {};
  GTile cur;
  fetch(0, cur);
  for (int kc = 0; kc < K; kc += 32) {
    stage(cur);
    GTile nxt;
    if (kc + 32 < K) fetch(kc + 32, nxt);   // overlap HBM with wmma below
    __syncthreads();
    // ---- fragment loads: two contiguous 8-half runs each (ds_load_b128) ----
    // A 16x32 layout: lane half 'hi' holds K = hi*8 + (0..7) and 16 + hi*8 + (0..7)
    v8h a0 = *(const v8h*)&Alds[ml][hi * 8];
    v8h a1 = *(const v8h*)&Alds[ml][16 + hi * 8];
    // B 32x16 layout: lane half 'hi' holds K = hi*8 + (0..7) and 16 + hi*8 + (0..7)
    v8h b0 = *(const v8h*)&Blds[lane16][hi * 8];
    v8h b1 = *(const v8h*)&Blds[lane16][16 + hi * 8];
    v16h av = __builtin_shufflevector(a0, a1, 0,1,2,3,4,5,6,7,8,9,10,11,12,13,14,15);
    v16h bv = __builtin_shufflevector(b0, b1, 0,1,2,3,4,5,6,7,8,9,10,11,12,13,14,15);
    acc = __builtin_amdgcn_wmma_f32_16x16x32_f16(
        false, av, false, bv, (short)0, acc, false, false);
    __syncthreads();
    cur = nxt;
  }
  // ---- store: C/D layout VGPR r -> M = r (+8 for hi lanes) ----
#pragma unroll
  for (int r = 0; r < 8; ++r) {
    int row = gm + wave * 16 + r + hi * 8;
    int col = gn + lane16;
    if (row < M) {
      float v = acc[r];
      if (bias) v += bias[col];
      if (act == 1) v = eluf(v);
      C[(long long)row * ldc + col] = v;
    }
  }
}

// ---------- zero-padded W1 copy: [257,256] -> [288,256] ----------
__global__ void pad_w1(const float* __restrict__ W1, float* __restrict__ W1p) {
  int idx = blockIdx.x * blockDim.x + threadIdx.x;
  if (idx >= KPAD * 256) return;
  int row = idx >> 8;
  W1p[idx] = (row < 257) ? W1[idx] : 0.f;
}

// ---------- edge attention bias: biasE[e,h] = edge_attr[e,:] @ We_l[:,h] ----------
__global__ void edge_bias(const float* __restrict__ ea, const float* __restrict__ Wel,
                          float* __restrict__ biasE, int E) {
  long long idx = (long long)blockIdx.x * blockDim.x + threadIdx.x;
  if (idx >= (long long)E * HEADS) return;
  int h = (int)(idx & 7);
  long long e = idx >> 3;
  float acc = 0.f;
#pragma unroll
  for (int j = 0; j < EDIM; ++j) acc += ea[e * EDIM + j] * Wel[j * HEADS + h];
  biasE[idx] = acc;
}

// ---------- hop phase A: logits + segment max ----------
__global__ void hop_logits(const float* __restrict__ hh,
                           const int* __restrict__ src, const int* __restrict__ dst,
                           const float* __restrict__ geom, const float* __restrict__ a,
                           const float* __restrict__ bias,
                           float* __restrict__ logits, unsigned* __restrict__ segmax, int M) {
  long long idx = (long long)blockIdx.x * blockDim.x + threadIdx.x;
  if (idx >= (long long)M * HEADS) return;
  int h = (int)(idx & 7);
  long long m = idx >> 3;
  int s = src[m], dv = dst[m];
  const float*  ah   = a + h * (2 * DH + 1);
  const float4* qrow = (const float4*)(hh + (long long)dv * HID + h * DH);
  const float4* krow = (const float4*)(hh + (long long)s  * HID + h * DH);
  float lg = 0.f;
#pragma unroll
  for (int v = 0; v < 4; ++v) {
    float4 qv = qrow[v], kv = krow[v];
    const float* aq = ah + 4 * v;
    const float* ak = ah + DH + 4 * v;
    lg += qv.x*aq[0] + qv.y*aq[1] + qv.z*aq[2] + qv.w*aq[3];
    lg += kv.x*ak[0] + kv.y*ak[1] + kv.z*ak[2] + kv.w*ak[3];
  }
  lg += geom[m] * ah[2 * DH];
  if (bias) lg += bias[m * HEADS + h];
  lg = lg > 0.f ? lg : 0.2f * lg;           // leaky relu
  logits[idx] = lg;
  atomicMax(&segmax[(long long)dv * HEADS + h], fenc(lg));
}

// ---------- hop phase B: exp(logit - max) + segment sum ----------
__global__ void hop_exp(float* __restrict__ logits, const int* __restrict__ dst,
                        const unsigned* __restrict__ segmax, float* __restrict__ segsum, int M) {
  long long idx = (long long)blockIdx.x * blockDim.x + threadIdx.x;
  if (idx >= (long long)M * HEADS) return;
  int h = (int)(idx & 7);
  long long m = idx >> 3;
  int dv = dst[m];
  float mx = fdec(segmax[(long long)dv * HEADS + h]);
  if (!(mx == mx)) mx = 0.f;                 // empty segment encodes NaN
  float e = __expf(logits[idx] - mx);
  logits[idx] = e;
  atomicAdd(&segsum[(long long)dv * HEADS + h], e);
}

// ---------- hop phase C: agg[dst] += k * alpha ----------
__global__ void hop_agg(const float* __restrict__ hh,
                        const int* __restrict__ src, const int* __restrict__ dst,
                        const float* __restrict__ logits, const float* __restrict__ segsum,
                        float* __restrict__ agg, int M) {
  long long idx = (long long)blockIdx.x * blockDim.x + threadIdx.x;
  if (idx >= (long long)M * HEADS) return;
  int h = (int)(idx & 7);
  long long m = idx >> 3;
  int s = src[m], dv = dst[m];
  float alpha = logits[idx] / (segsum[(long long)dv * HEADS + h] + 1e-16f);
  const float4* krow = (const float4*)(hh + (long long)s * HID + h * DH);
  float* arow = agg + (long long)dv * HID + h * DH;
#pragma unroll
  for (int v = 0; v < 4; ++v) {
    float4 kv = krow[v];
    atomicAdd(&arow[4*v + 0], kv.x * alpha);
    atomicAdd(&arow[4*v + 1], kv.y * alpha);
    atomicAdd(&arow[4*v + 2], kv.z * alpha);
    atomicAdd(&arow[4*v + 3], kv.w * alpha);
  }
}

// ---------- h = elu(hl + agg) ----------
__global__ void elu_combine(const float* __restrict__ hl, const float* __restrict__ agg,
                            float* __restrict__ h, int N) {
  long long idx = (long long)blockIdx.x * blockDim.x + threadIdx.x;
  if (idx >= (long long)N * HID) return;
  h[idx] = eluf(hl[idx] + agg[idx]);
}

// ---------- pooled readout scatter ----------
__global__ void pool_scatter(const float* __restrict__ h, const int* __restrict__ batch,
                             float* __restrict__ s, unsigned* __restrict__ mx, int N) {
  long long idx = (long long)blockIdx.x * blockDim.x + threadIdx.x;
  if (idx >= (long long)N * HID) return;
  int c = (int)(idx & (HID - 1));
  long long n = idx >> 7;
  int g = batch[n];
  float v = h[idx];
  atomicAdd(&s[(long long)g * HID + c], v);
  atomicMax(&mx[(long long)g * HID + c], fenc(v));
}

// ---------- build mol[G,288] = [sum(128), max(128), temp(1), zeros(31)] ----------
__global__ void mol_build(const float* __restrict__ s, const unsigned* __restrict__ mx,
                          const float* __restrict__ temps, float* __restrict__ mol, int G) {
  long long idx = (long long)blockIdx.x * blockDim.x + threadIdx.x;
  if (idx >= (long long)G * KPAD) return;
  int c = (int)(idx % KPAD);
  long long g = idx / KPAD;
  float v;
  if (c < HID) {
    v = s[g * HID + c];
  } else if (c < 2 * HID) {
    v = fdec(mx[g * HID + (c - HID)]);
    if (!(v == v)) v = 0.f;                  // non-finite max -> 0
  } else if (c == 2 * HID) {
    v = temps[g];
  } else {
    v = 0.f;                                 // zero K-pad
  }
  mol[idx] = v;
}

// ---------- final head: out[g] = hdn2[g,:] @ W3 + b3 ----------
__global__ void final_head(const float* __restrict__ hdn2, const float* __restrict__ W3,
                           const float* __restrict__ b3, float* __restrict__ out, int G) {
  int g = blockIdx.x * blockDim.x + threadIdx.x;
  if (g >= G) return;
  const float4* row = (const float4*)(hdn2 + (long long)g * 256);
  const float4* w   = (const float4*)W3;
  float acc = 0.f;
#pragma unroll 4
  for (int j = 0; j < 64; ++j) {
    float4 r = row[j], wv = w[j];
    acc += r.x*wv.x + r.y*wv.y + r.z*wv.z + r.w*wv.w;
  }
  out[g] = acc + b3[0];
}

// ---------- host launch ----------
extern "C" void kernel_launch(void* const* d_in, const int* in_sizes, int n_in,
                              void* d_out, int out_size, void* d_ws, size_t ws_size,
                              hipStream_t stream) {
  (void)n_in; (void)out_size; (void)ws_size;
  const float* x          = (const float*)d_in[0];
  const float* pos        = (const float*)d_in[1];
  const float* edge_attr  = (const float*)d_in[2];
  const float* temps      = (const float*)d_in[3];
  const int*   edge_index = (const int*)d_in[4];
  const int*   triple_idx = (const int*)d_in[5];
  const int*   quadra_idx = (const int*)d_in[6];
  const int*   batch      = (const int*)d_in[7];
  const float* W   = (const float*)d_in[8];
  const float* a_e = (const float*)d_in[9];
  const float* a_t = (const float*)d_in[10];
  const float* a_q = (const float*)d_in[11];
  const float* We  = (const float*)d_in[12];
  const float* W1  = (const float*)d_in[13];
  const float* b1  = (const float*)d_in[14];
  const float* W2  = (const float*)d_in[15];
  const float* b2  = (const float*)d_in[16];
  const float* W3  = (const float*)d_in[17];
  const float* b3  = (const float*)d_in[18];

  const int N = in_sizes[0] / HID;
  const int E = in_sizes[2] / EDIM;
  const int T = in_sizes[5] / 3;
  const int Q = in_sizes[6] / 4;
  const int G = in_sizes[3];

  char* ws = (char*)d_ws;
  size_t off = 0;
  auto carve = [&](size_t bytes) -> char* {
    char* p = ws + off;
    off = (off + bytes + 255) & ~(size_t)255;
    return p;
  };
  float*    d_e    = (float*)carve((size_t)E * 4);
  float*    cos_a  = (float*)carve((size_t)T * 4);
  float*    cos_d  = (float*)carve((size_t)Q * 4);
  float*    biasE  = (float*)carve((size_t)E * HEADS * 4);
  float*    hbuf   = (float*)carve((size_t)N * HID * 4);
  float*    hl     = (float*)carve((size_t)N * HID * 4);
  float*    agg    = (float*)carve((size_t)N * HID * 4);
  unsigned* segmax = (unsigned*)carve((size_t)N * HEADS * 4);
  float*    segsum = (float*)carve((size_t)N * HEADS * 4);
  size_t maxM = (size_t)(E > T ? E : T); if ((size_t)Q > maxM) maxM = Q;
  char* scratch = carve(maxM * HEADS * 4);       // logits; reused post-hops for readout
  float*    logits  = (float*)scratch;
  float*    s_pool  = (float*)scratch;           // overlay: used only after all hops
  unsigned* mx_pool = (unsigned*)(s_pool + (size_t)G * HID);
  float*    mol   = (float*)(mx_pool + (size_t)G * HID);
  float*    hdn1  = mol + (size_t)G * KPAD;
  float*    hdn2  = hdn1 + (size_t)G * 256;
  float*    W1pad = hdn2 + (size_t)G * 256;

  const int bs = 256;
  edge_geom  <<<(E + bs - 1) / bs, bs, 0, stream>>>(pos, edge_index, d_e, E);
  triple_geom<<<(T + bs - 1) / bs, bs, 0, stream>>>(pos, triple_idx, cos_a, T);
  quad_geom  <<<(Q + bs - 1) / bs, bs, 0, stream>>>(pos, quadra_idx, cos_d, Q);

  const float* hcur = x;
  for (int l = 0; l < LAYERS; ++l) {
    dim3 g1(HID / 16, (N + TM - 1) / TM);
    wmma_gemm<<<g1, 128, 0, stream>>>(hcur, HID, W + (size_t)l * HID * HID, HID,
                                      hl, HID, N, HID, nullptr, 0);
    long long EH = (long long)E * HEADS;
    edge_bias<<<(unsigned)((EH + bs - 1) / bs), bs, 0, stream>>>(
        edge_attr, We + (size_t)l * EDIM * HEADS, biasE, E);
    hipMemsetAsync(agg, 0, (size_t)N * HID * 4, stream);

    struct Hop { const int* src; const int* dst; const float* geom; const float* a; const float* bias; int M; };
    Hop hops[3] = {
      { edge_index + E,              edge_index, d_e,   a_e + (size_t)l * HEADS * 33, biasE,   E },
      { triple_idx + 2 * (size_t)T,  triple_idx, cos_a, a_t + (size_t)l * HEADS * 33, nullptr, T },
      { quadra_idx + 3 * (size_t)Q,  quadra_idx, cos_d, a_q + (size_t)l * HEADS * 33, nullptr, Q },
    };
    for (int hp = 0; hp < 3; ++hp) {
      hipMemsetAsync(segmax, 0, (size_t)N * HEADS * 4, stream);
      hipMemsetAsync(segsum, 0, (size_t)N * HEADS * 4, stream);
      long long MH = (long long)hops[hp].M * HEADS;
      unsigned nb = (unsigned)((MH + bs - 1) / bs);
      hop_logits<<<nb, bs, 0, stream>>>(hl, hops[hp].src, hops[hp].dst, hops[hp].geom,
                                        hops[hp].a, hops[hp].bias, logits, segmax, hops[hp].M);
      hop_exp   <<<nb, bs, 0, stream>>>(logits, hops[hp].dst, segmax, segsum, hops[hp].M);
      hop_agg   <<<nb, bs, 0, stream>>>(hl, hops[hp].src, hops[hp].dst, logits, segsum,
                                        agg, hops[hp].M);
    }
    long long NC = (long long)N * HID;
    elu_combine<<<(unsigned)((NC + bs - 1) / bs), bs, 0, stream>>>(hl, agg, hbuf, N);
    hcur = hbuf;
  }

  // readout
  hipMemsetAsync(s_pool, 0, (size_t)G * HID * 4, stream);
  hipMemsetAsync(mx_pool, 0, (size_t)G * HID * 4, stream);
  long long NC = (long long)N * HID;
  pool_scatter<<<(unsigned)((NC + bs - 1) / bs), bs, 0, stream>>>(hbuf, batch, s_pool, mx_pool, N);
  long long GM = (long long)G * KPAD;
  mol_build<<<(unsigned)((GM + bs - 1) / bs), bs, 0, stream>>>(s_pool, mx_pool, temps, mol, G);
  pad_w1<<<(KPAD * 256 + bs - 1) / bs, bs, 0, stream>>>(W1, W1pad);

  dim3 g2(256 / 16, G / TM);
  wmma_gemm<<<g2, 128, 0, stream>>>(mol,  KPAD, W1pad, 256, hdn1, 256, G, KPAD, b1, 1);
  wmma_gemm<<<g2, 128, 0, stream>>>(hdn1, 256,  W2,    256, hdn2, 256, G, 256,  b2, 1);
  final_head<<<(G + bs - 1) / bs, bs, 0, stream>>>(hdn2, W3, b3, (float*)d_out, G);
}
